// EntAttentionLayer_18124761989312
// MI455X (gfx1250) — matrix operand order repaired and checked by reading, not compiled, not run on hardware
//
#include <hip/hip_runtime.h>

// ---------------------------------------------------------------------------
// CDNA5 (gfx1250) BERT-style entity-attention layer.
// All matmuls: v_wmma_f32_16x16x32_bf16, fp32 accumulate, wave32.
// All GEMM A-operands are bf16 in memory (producers emit bf16 copies);
// fp32 kept only for pre-LN sums / residuals / final output.
// GEMM is LDS double-buffered (1 barrier per K-step, global loads overlap
// WMMA). Softmax row reductions use v_permlane16 (VALU) instead of
// ds_bpermute, keeping the DS pipe free for WMMA fragment loads.
// ---------------------------------------------------------------------------

typedef __attribute__((ext_vector_type(16))) __bf16 bf16x16;
typedef __attribute__((ext_vector_type(8)))  float  floatx8;

struct FragBF16 { union { bf16x16 v; uint4 q[2]; }; };
union U4BF { uint4 q; __bf16 h[8]; };

// A-matrix fragment (16x32 bf16): lane holds row (lane&15); element j=0..7 is
// K = half*8 + j, j=8..15 is K = 16 + half*8 + (j-8)  (ISA 7.12.2).
__device__ inline FragBF16 frag_a(const __bf16* rowbase, int hlf) {
  FragBF16 f;
  const char* p = (const char*)rowbase + hlf * 16;     // half*8 elems * 2B
  f.q[0] = *(const uint4*)(p);
  f.q[1] = *(const uint4*)(p + 32);                    // +16 elems
  return f;
}
// B-matrix fragment (32x16 bf16): lane holds column (lane&15); element j=0..7 is
// K = half*8 + j, j=8..15 is K = 16 + half*8 + (j-8) (same as A).
__device__ inline FragBF16 frag_b(const __bf16* colbase, int hlf) {
  FragBF16 f;
  const char* p = (const char*)colbase + hlf * 16;     // half*8 elems * 2B
  f.q[0] = *(const uint4*)(p);
  f.q[1] = *(const uint4*)(p + 32);
  return f;
}
__device__ inline floatx8 wmma_bf16(const FragBF16& a, const FragBF16& b, floatx8 c) {
  return __builtin_amdgcn_wmma_f32_16x16x32_bf16(false, a.v, false, b.v,
                                                 (short)0, c, false, false);
}
__device__ inline floatx8 fzero8() {
  floatx8 z = {0.f,0.f,0.f,0.f,0.f,0.f,0.f,0.f};
  return z;
}

// --------------------- 16-lane xor-butterfly reductions --------------------
// v_permlane16_b32: per-row-of-16 arbitrary permute, selector nibbles in
// (sel_lo, sel_hi). Pure VALU (no DS pipe / dscnt waits).
#if defined(__has_builtin)
#if __has_builtin(__builtin_amdgcn_permlane16)
#define HAVE_PERMLANE16 1
#endif
#endif

#ifdef HAVE_PERMLANE16
__device__ inline float permrow16(float x, unsigned lo, unsigned hi) {
  unsigned u = (unsigned)__float_as_int(x);
  unsigned r = __builtin_amdgcn_permlane16(u, u, lo, hi, false, false);
  return __int_as_float((int)r);
}
#endif

__device__ inline float rowmax16(float x) {
#ifdef HAVE_PERMLANE16
  x = fmaxf(x, permrow16(x, 0x67452301u, 0xEFCDAB89u));  // xor 1
  x = fmaxf(x, permrow16(x, 0x54761032u, 0xDCFE98BAu));  // xor 2
  x = fmaxf(x, permrow16(x, 0x32107654u, 0xBA98FEDCu));  // xor 4
  x = fmaxf(x, permrow16(x, 0xFEDCBA98u, 0x76543210u));  // xor 8
#else
#pragma unroll
  for (int m = 1; m < 16; m <<= 1) x = fmaxf(x, __shfl_xor(x, m, 32));
#endif
  return x;
}
__device__ inline float rowsum16(float x) {
#ifdef HAVE_PERMLANE16
  x += permrow16(x, 0x67452301u, 0xEFCDAB89u);
  x += permrow16(x, 0x54761032u, 0xDCFE98BAu);
  x += permrow16(x, 0x32107654u, 0xBA98FEDCu);
  x += permrow16(x, 0xFEDCBA98u, 0x76543210u);
#else
#pragma unroll
  for (int m = 1; m < 16; m <<= 1) x += __shfl_xor(x, m, 32);
#endif
  return x;
}

// ---------------------------------------------------------------------------
// GEMM: out[M,N] = epilogue(A[M,K](bf16) @ W[K,N](f32->bf16) + bias)
// EPI: 0 = bias, 1 = bias + exact GELU, 2 = bias + residual add (f32 out)
// OUTB: write bf16 output (feeds another GEMM/attention) else f32.
// Block: 256 thr (8 waves, 4x2), tile 128x128, BK=32, double-buffered LDS.
// LDS stride 40 bf16 = 80B: rows 16B-aligned, b128 fragment reads bank-clean.
// ---------------------------------------------------------------------------
#define GEMM_BK 32
#define GEMM_LD 40

#define GEMM_GLOAD(K0) do {                                                   \
    if (aValid) {                                                             \
      a0r = *(const uint4*)(aSrc + (K0));                                     \
      a1r = *(const uint4*)(aSrc + (K0) + 8);                                 \
    } else { a0r = uint4{0u,0u,0u,0u}; a1r = uint4{0u,0u,0u,0u}; }            \
    const float* _s = bSrc + (size_t)(K0) * N;                                \
    w0r = *(const float4*)(_s);      w1r = *(const float4*)(_s + 4);          \
    w2r = *(const float4*)(_s + 8);  w3r = *(const float4*)(_s + 12);         \
  } while (0)

#define GEMM_LSTORE(BUF) do {                                                 \
    __bf16* _da = &sA[BUF][arow * GEMM_LD + acs];                             \
    *(uint4*)(_da) = a0r; *(uint4*)(_da + 8) = a1r;                           \
    __bf16* _db = &sB[BUF][0];                                                \
    _db[(bns+ 0)*GEMM_LD + bk] = (__bf16)w0r.x;                               \
    _db[(bns+ 1)*GEMM_LD + bk] = (__bf16)w0r.y;                               \
    _db[(bns+ 2)*GEMM_LD + bk] = (__bf16)w0r.z;                               \
    _db[(bns+ 3)*GEMM_LD + bk] = (__bf16)w0r.w;                               \
    _db[(bns+ 4)*GEMM_LD + bk] = (__bf16)w1r.x;                               \
    _db[(bns+ 5)*GEMM_LD + bk] = (__bf16)w1r.y;                               \
    _db[(bns+ 6)*GEMM_LD + bk] = (__bf16)w1r.z;                               \
    _db[(bns+ 7)*GEMM_LD + bk] = (__bf16)w1r.w;                               \
    _db[(bns+ 8)*GEMM_LD + bk] = (__bf16)w2r.x;                               \
    _db[(bns+ 9)*GEMM_LD + bk] = (__bf16)w2r.y;                               \
    _db[(bns+10)*GEMM_LD + bk] = (__bf16)w2r.z;                               \
    _db[(bns+11)*GEMM_LD + bk] = (__bf16)w2r.w;                               \
    _db[(bns+12)*GEMM_LD + bk] = (__bf16)w3r.x;                               \
    _db[(bns+13)*GEMM_LD + bk] = (__bf16)w3r.y;                               \
    _db[(bns+14)*GEMM_LD + bk] = (__bf16)w3r.z;                               \
    _db[(bns+15)*GEMM_LD + bk] = (__bf16)w3r.w;                               \
  } while (0)

template<int EPI, bool OUTB>
__global__ __launch_bounds__(256) void gemm_kernel(
    const __bf16* __restrict__ A, const float* __restrict__ W,
    const float* __restrict__ bias, const float* __restrict__ res,
    float* __restrict__ C, __bf16* __restrict__ Cb, int M, int N, int K)
{
  __shared__ __align__(16) __bf16 sA[2][128 * GEMM_LD];
  __shared__ __align__(16) __bf16 sB[2][128 * GEMM_LD];   // W tile, [n][k]

  const int tid = threadIdx.x, lane = tid & 31, wave = tid >> 5;
  const int hlf = lane >> 4, ln = lane & 15;
  const int m0 = blockIdx.x * 128, n0 = blockIdx.y * 128;
  const int wm = (wave >> 1) * 32, wn = (wave & 1) * 64;

  floatx8 acc[2][4];
#pragma unroll
  for (int i = 0; i < 2; ++i)
#pragma unroll
    for (int j = 0; j < 4; ++j) acc[i][j] = fzero8();

  const int arow = tid >> 1, acs = (tid & 1) * 16;   // A: 2 thr/row, 16 bf16
  const int bk = tid >> 3, bns = (tid & 7) * 16;     // W: 8 thr/row, 16 f32
  const __bf16* aSrc = A + (size_t)(m0 + arow) * K + acs;
  const float*  bSrc = W + (size_t)bk * N + n0 + bns;
  const bool aValid = (m0 + arow) < M;

  uint4 a0r, a1r; float4 w0r, w1r, w2r, w3r;
  GEMM_GLOAD(0);
  GEMM_LSTORE(0);
  int cur = 0;
  for (int k0 = 0; k0 < K; k0 += GEMM_BK) {
    __syncthreads();                                  // buf[cur] ready
    const bool nxt = (k0 + GEMM_BK) < K;
    if (nxt) GEMM_GLOAD(k0 + GEMM_BK);                // overlap with WMMA

    const __bf16* sAc = sA[cur];
    const __bf16* sBc = sB[cur];
    FragBF16 fa0 = frag_a(sAc + (wm      + ln) * GEMM_LD, hlf);
    FragBF16 fa1 = frag_a(sAc + (wm + 16 + ln) * GEMM_LD, hlf);
#pragma unroll
    for (int fn = 0; fn < 4; ++fn) {
      FragBF16 fb = frag_b(sBc + (wn + fn*16 + ln) * GEMM_LD, hlf);
      acc[0][fn] = wmma_bf16(fa0, fb, acc[0][fn]);
      acc[1][fn] = wmma_bf16(fa1, fb, acc[1][fn]);
    }
    if (nxt) GEMM_LSTORE(cur ^ 1);                    // other buffer: no race
    cur ^= 1;
  }

#pragma unroll
  for (int fm = 0; fm < 2; ++fm) {
#pragma unroll
    for (int fn = 0; fn < 4; ++fn) {
      const int gn = n0 + wn + fn*16 + ln;
      const float bv = bias[gn];
#pragma unroll
      for (int r = 0; r < 8; ++r) {
        const int gm = m0 + wm + fm*16 + hlf*8 + r;   // C/D layout (ISA 7.12.2)
        if (gm < M) {
          float v = acc[fm][fn][r] + bv;
          if (EPI == 1) v = 0.5f * v * (1.0f + erff(v * 0.70710678118654752f));
          if (OUTB) {
            Cb[(size_t)gm * N + gn] = (__bf16)v;
          } else {
            if (EPI == 2) v += res[(size_t)gm * N + gn];
            C[(size_t)gm * N + gn] = v;
          }
        }
      }
    }
  }
}

// ---------------------------------------------------------------------------
// Flash self-attention with ADDITIVE band bonus (+1 inside |q-k|<=range).
// Full softmax over S keys (mask is additive, not sparsifying).
// Block: 128 thr (4 waves), 64 queries / block, wave owns 16 query rows.
// bf16 in / bf16 out; row stats via v_permlane16 butterflies.
// ---------------------------------------------------------------------------
#define ATT_LD 72

__global__ __launch_bounds__(128) void flash_self_attn_kernel(
    const __bf16* __restrict__ Q, const __bf16* __restrict__ K,
    const __bf16* __restrict__ V, __bf16* __restrict__ O,
    const int* __restrict__ ent_range_p, int B, int S, int NH, int HD)
{
  __shared__ __align__(16) __bf16 sQ [64 * ATT_LD];
  __shared__ __align__(16) __bf16 sK [64 * ATT_LD];   // [key][d]
  __shared__ __align__(16) __bf16 sVt[64 * ATT_LD];   // [d][key]
  __shared__ __align__(16) __bf16 sP [64 * ATT_LD];   // [q][key]

  const int tid = threadIdx.x, lane = tid & 31, wave = tid >> 5;
  const int hlf = lane >> 4, ln = lane & 15;
  const int q0 = blockIdx.x * 64;
  const int bh = blockIdx.y, bb = bh / NH, h = bh % NH;
  const int H = NH * HD;
  const int ent_range = *ent_range_p;
  const float scale = 0.125f;   // 1/sqrt(64)

  const int row = tid >> 1, cb = (tid & 1) * 32;
  {   // Q tile (64x64 bf16) -> LDS, straight b128 copies
    const __bf16* src = Q + ((size_t)bb * S + q0 + row) * H + h * HD + cb;
    __bf16* dst = sQ + row * ATT_LD + cb;
#pragma unroll
    for (int i = 0; i < 4; ++i) *(uint4*)(dst + i*8) = *(const uint4*)(src + i*8);
  }
  __syncthreads();
  const int qr = wave * 16 + ln;
  FragBF16 aq0 = frag_a(sQ + qr * ATT_LD + 0,  hlf);
  FragBF16 aq1 = frag_a(sQ + qr * ATT_LD + 32, hlf);

  floatx8 oAcc[4];
#pragma unroll
  for (int dn = 0; dn < 4; ++dn) oAcc[dn] = fzero8();
  float mI[8], lI[8];
#pragma unroll
  for (int r = 0; r < 8; ++r) { mI[r] = -1e30f; lI[r] = 0.f; }

  const int nKB = S / 64;
  for (int kb = 0; kb < nKB; ++kb) {
    __syncthreads();
    {   // K tile (row-major copy) and V tile (transposed) -> LDS
      const __bf16* srck = K + ((size_t)bb * S + kb*64 + row) * H + h * HD + cb;
      const __bf16* srcv = V + ((size_t)bb * S + kb*64 + row) * H + h * HD + cb;
      __bf16* dstk = sK + row * ATT_LD + cb;
#pragma unroll
      for (int i = 0; i < 4; ++i) {
        *(uint4*)(dstk + i*8) = *(const uint4*)(srck + i*8);
        U4BF u; u.q = *(const uint4*)(srcv + i*8);
#pragma unroll
        for (int e = 0; e < 8; ++e) sVt[(cb + i*8 + e) * ATT_LD + row] = u.h[e];
      }
    }
    __syncthreads();

    // S = Q @ K^T  (4 key-frags x 2 d-chunks)
    floatx8 sAcc[4];
#pragma unroll
    for (int kn = 0; kn < 4; ++kn) {
      FragBF16 b0 = frag_b(sK + (kn*16 + ln) * ATT_LD + 0,  hlf);
      FragBF16 b1 = frag_b(sK + (kn*16 + ln) * ATT_LD + 32, hlf);
      floatx8 a = fzero8();
      a = wmma_bf16(aq0, b0, a);
      a = wmma_bf16(aq1, b1, a);
      sAcc[kn] = a;
    }

    // scale + additive band bonus; per-row max
    const int qgBase = q0 + wave*16 + hlf*8;
    float rmax[8];
#pragma unroll
    for (int r = 0; r < 8; ++r) rmax[r] = -1e30f;
#pragma unroll
    for (int kn = 0; kn < 4; ++kn) {
      const int kg = kb*64 + kn*16 + ln;
#pragma unroll
      for (int r = 0; r < 8; ++r) {
        float v = sAcc[kn][r] * scale;
        int dq = qgBase + r - kg; dq = dq < 0 ? -dq : dq;
        v += (dq <= ent_range) ? 1.0f : 0.0f;
        sAcc[kn][r] = v;
        rmax[r] = fmaxf(rmax[r], v);
      }
    }
#pragma unroll
    for (int r = 0; r < 8; ++r) rmax[r] = rowmax16(rmax[r]);

    float fac[8], rsum[8];
#pragma unroll
    for (int r = 0; r < 8; ++r) {
      float mn = fmaxf(mI[r], rmax[r]);
      fac[r] = __expf(mI[r] - mn);
      mI[r] = mn; rsum[r] = 0.f;
    }
#pragma unroll
    for (int kn = 0; kn < 4; ++kn)
#pragma unroll
      for (int r = 0; r < 8; ++r) {
        float p = __expf(sAcc[kn][r] - mI[r]);
        sAcc[kn][r] = p; rsum[r] += p;
      }
#pragma unroll
    for (int r = 0; r < 8; ++r) rsum[r] = rowsum16(rsum[r]);
#pragma unroll
    for (int r = 0; r < 8; ++r) lI[r] = lI[r] * fac[r] + rsum[r];
#pragma unroll
    for (int dn = 0; dn < 4; ++dn)
#pragma unroll
      for (int r = 0; r < 8; ++r) oAcc[dn][r] *= fac[r];

    // P -> LDS (accumulator layout -> A-fragment layout)
#pragma unroll
    for (int kn = 0; kn < 4; ++kn)
#pragma unroll
      for (int r = 0; r < 8; ++r)
        sP[(wave*16 + hlf*8 + r) * ATT_LD + kn*16 + ln] = (__bf16)sAcc[kn][r];
    __syncthreads();

    FragBF16 p0 = frag_a(sP + qr * ATT_LD + 0,  hlf);
    FragBF16 p1 = frag_a(sP + qr * ATT_LD + 32, hlf);
#pragma unroll
    for (int dn = 0; dn < 4; ++dn) {
      FragBF16 v0 = frag_b(sVt + (dn*16 + ln) * ATT_LD + 0,  hlf);
      FragBF16 v1 = frag_b(sVt + (dn*16 + ln) * ATT_LD + 32, hlf);
      oAcc[dn] = wmma_bf16(p0, v0, oAcc[dn]);
      oAcc[dn] = wmma_bf16(p1, v1, oAcc[dn]);
    }
  }

  float inv[8];
#pragma unroll
  for (int r = 0; r < 8; ++r) inv[r] = 1.0f / lI[r];
#pragma unroll
  for (int dn = 0; dn < 4; ++dn)
#pragma unroll
    for (int r = 0; r < 8; ++r) {
      const int m = wave*16 + hlf*8 + r;
      O[((size_t)bb * S + q0 + m) * H + h * HD + dn*16 + ln] =
          (__bf16)(oAcc[dn][r] * inv[r]);
    }
}

// ---------------------------------------------------------------------------
// Cross-attention onto T=50 tag keys: single padded 64-key tile, keys >= T
// masked to -1e30, direct softmax. K/V are [T,H] bf16, batch-invariant.
// ---------------------------------------------------------------------------
__global__ __launch_bounds__(128) void cross_attn_kernel(
    const __bf16* __restrict__ Q, const __bf16* __restrict__ Kc,
    const __bf16* __restrict__ Vc, __bf16* __restrict__ O,
    int B, int S, int NH, int HD, int T)
{
  __shared__ __align__(16) __bf16 sQ [64 * ATT_LD];
  __shared__ __align__(16) __bf16 sK [64 * ATT_LD];
  __shared__ __align__(16) __bf16 sVt[64 * ATT_LD];
  __shared__ __align__(16) __bf16 sP [64 * ATT_LD];

  const int tid = threadIdx.x, lane = tid & 31, wave = tid >> 5;
  const int hlf = lane >> 4, ln = lane & 15;
  const int q0 = blockIdx.x * 64;
  const int bh = blockIdx.y, bb = bh / NH, h = bh % NH;
  const int H = NH * HD;
  const float scale = 0.125f;

  const int row = tid >> 1, cb = (tid & 1) * 32;
  {
    const __bf16* src = Q + ((size_t)bb * S + q0 + row) * H + h * HD + cb;
    __bf16* dst = sQ + row * ATT_LD + cb;
#pragma unroll
    for (int i = 0; i < 4; ++i) *(uint4*)(dst + i*8) = *(const uint4*)(src + i*8);

    __bf16* dstk = sK + row * ATT_LD + cb;
    if (row < T) {
      const __bf16* srck = Kc + (size_t)row * H + h * HD + cb;
      const __bf16* srcv = Vc + (size_t)row * H + h * HD + cb;
#pragma unroll
      for (int i = 0; i < 4; ++i) {
        *(uint4*)(dstk + i*8) = *(const uint4*)(srck + i*8);
        U4BF u; u.q = *(const uint4*)(srcv + i*8);
#pragma unroll
        for (int e = 0; e < 8; ++e) sVt[(cb + i*8 + e) * ATT_LD + row] = u.h[e];
      }
    } else {
      const uint4 z = uint4{0u,0u,0u,0u};
#pragma unroll
      for (int i = 0; i < 4; ++i) *(uint4*)(dstk + i*8) = z;
#pragma unroll
      for (int i = 0; i < 32; ++i) sVt[(cb + i) * ATT_LD + row] = (__bf16)0.f;
    }
  }
  __syncthreads();

  const int qr = wave * 16 + ln;
  FragBF16 aq0 = frag_a(sQ + qr * ATT_LD + 0,  hlf);
  FragBF16 aq1 = frag_a(sQ + qr * ATT_LD + 32, hlf);

  floatx8 sAcc[4];
#pragma unroll
  for (int kn = 0; kn < 4; ++kn) {
    FragBF16 b0 = frag_b(sK + (kn*16 + ln) * ATT_LD + 0,  hlf);
    FragBF16 b1 = frag_b(sK + (kn*16 + ln) * ATT_LD + 32, hlf);
    floatx8 a = fzero8();
    a = wmma_bf16(aq0, b0, a);
    a = wmma_bf16(aq1, b1, a);
    sAcc[kn] = a;
  }

  float rmax[8];
#pragma unroll
  for (int r = 0; r < 8; ++r) rmax[r] = -1e30f;
#pragma unroll
  for (int kn = 0; kn < 4; ++kn) {
    const int kg = kn*16 + ln;
#pragma unroll
    for (int r = 0; r < 8; ++r) {
      float v = (kg < T) ? sAcc[kn][r] * scale : -1e30f;
      sAcc[kn][r] = v;
      rmax[r] = fmaxf(rmax[r], v);
    }
  }
#pragma unroll
  for (int r = 0; r < 8; ++r) rmax[r] = rowmax16(rmax[r]);

  float rsum[8];
#pragma unroll
  for (int r = 0; r < 8; ++r) rsum[r] = 0.f;
#pragma unroll
  for (int kn = 0; kn < 4; ++kn)
#pragma unroll
    for (int r = 0; r < 8; ++r) {
      float p = __expf(sAcc[kn][r] - rmax[r]);
      sAcc[kn][r] = p; rsum[r] += p;
    }
#pragma unroll
  for (int r = 0; r < 8; ++r) rsum[r] = rowsum16(rsum[r]);

#pragma unroll
  for (int kn = 0; kn < 4; ++kn)
#pragma unroll
    for (int r = 0; r < 8; ++r)
      sP[(wave*16 + hlf*8 + r) * ATT_LD + kn*16 + ln] = (__bf16)sAcc[kn][r];
  __syncthreads();

  FragBF16 p0 = frag_a(sP + qr * ATT_LD + 0,  hlf);
  FragBF16 p1 = frag_a(sP + qr * ATT_LD + 32, hlf);
  floatx8 oAcc[4];
#pragma unroll
  for (int dn = 0; dn < 4; ++dn) {
    FragBF16 v0 = frag_b(sVt + (dn*16 + ln) * ATT_LD + 0,  hlf);
    FragBF16 v1 = frag_b(sVt + (dn*16 + ln) * ATT_LD + 32, hlf);
    floatx8 a = fzero8();
    a = wmma_bf16(p0, v0, a);
    a = wmma_bf16(p1, v1, a);
    oAcc[dn] = a;
  }

  float inv[8];
#pragma unroll
  for (int r = 0; r < 8; ++r) inv[r] = 1.0f / rsum[r];
#pragma unroll
  for (int dn = 0; dn < 4; ++dn)
#pragma unroll
    for (int r = 0; r < 8; ++r) {
      const int m = wave*16 + hlf*8 + r;
      O[((size_t)bb * S + q0 + m) * H + h * HD + dn*16 + ln] =
          (__bf16)(oAcc[dn][r] * inv[r]);
    }
}

// ---------------------------------------------------------------------------
// Row LayerNorm; optional secondary bf16 output (feeds the next GEMM).
// ---------------------------------------------------------------------------
__global__ __launch_bounds__(256) void layernorm_kernel(
    const float* __restrict__ X, const float* __restrict__ g,
    const float* __restrict__ b, float* __restrict__ Y,
    __bf16* __restrict__ Yb, int H)
{
  const size_t rw = blockIdx.x;
  const float* x = X + rw * H;
  const int tid = threadIdx.x;
  float s = 0.f, s2 = 0.f;
  for (int i = tid; i < H; i += 256) { float v = x[i]; s += v; s2 += v * v; }
#pragma unroll
  for (int m = 16; m; m >>= 1) { s += __shfl_xor(s, m, 32); s2 += __shfl_xor(s2, m, 32); }
  __shared__ float rs[8], rs2[8];
  if ((tid & 31) == 0) { rs[tid >> 5] = s; rs2[tid >> 5] = s2; }
  __syncthreads();
  s = 0.f; s2 = 0.f;
#pragma unroll
  for (int w = 0; w < 8; ++w) { s += rs[w]; s2 += rs2[w]; }
  const float mean = s / H;
  const float var = s2 / H - mean * mean;
  const float inv = rsqrtf(var + 1e-12f);
  for (int i = tid; i < H; i += 256) {
    float v = (x[i] - mean) * inv * g[i] + b[i];
    if (Y)  Y[rw * H + i] = v;
    if (Yb) Yb[rw * H + i] = (__bf16)v;
  }
}

__global__ __launch_bounds__(256) void cvt_bf16_kernel(
    const float* __restrict__ X, __bf16* __restrict__ Y, int n4)
{
  const int i = blockIdx.x * 256 + threadIdx.x;
  if (i < n4) {
    float4 f = ((const float4*)X)[i];
    __bf16* y = Y + (size_t)i * 4;
    y[0] = (__bf16)f.x; y[1] = (__bf16)f.y; y[2] = (__bf16)f.z; y[3] = (__bf16)f.w;
  }
}

__global__ __launch_bounds__(256) void gather_tags_kernel(
    const float* __restrict__ emb, const int* __restrict__ ids,
    __bf16* __restrict__ tags, int H)
{
  const int t = blockIdx.x;
  const int id = ids[t];
  for (int i = threadIdx.x; i < H; i += 256)
    tags[(size_t)t * H + i] = (__bf16)emb[(size_t)id * H + i];
}

// ---------------------------------------------------------------------------
extern "C" void kernel_launch(void* const* d_in, const int* in_sizes, int n_in,
                              void* d_out, int out_size, void* d_ws, size_t ws_size,
                              hipStream_t stream) {
  (void)in_sizes; (void)n_in; (void)out_size; (void)ws_size;
  const float* x      = (const float*)d_in[0];
  const float* emb    = (const float*)d_in[1];
  const float* sq_w   = (const float*)d_in[2];  const float* sq_b = (const float*)d_in[3];
  const float* sk_w   = (const float*)d_in[4];  const float* sk_b = (const float*)d_in[5];
  const float* sv_w   = (const float*)d_in[6];  const float* sv_b = (const float*)d_in[7];
  const float* so_w   = (const float*)d_in[8];  const float* so_b = (const float*)d_in[9];
  const float* sln_g  = (const float*)d_in[10]; const float* sln_b = (const float*)d_in[11];
  const float* cq_w   = (const float*)d_in[12]; const float* cq_b = (const float*)d_in[13];
  const float* ck_w   = (const float*)d_in[14]; const float* ck_b = (const float*)d_in[15];
  const float* cv_w   = (const float*)d_in[16]; const float* cv_b = (const float*)d_in[17];
  const float* co_w   = (const float*)d_in[18]; const float* co_b = (const float*)d_in[19];
  const float* cln_g  = (const float*)d_in[20]; const float* cln_b = (const float*)d_in[21];
  const float* i_w    = (const float*)d_in[22]; const float* i_b  = (const float*)d_in[23];
  const float* o_w    = (const float*)d_in[24]; const float* o_b  = (const float*)d_in[25];
  const float* oln_g  = (const float*)d_in[26]; const float* oln_b = (const float*)d_in[27];
  const int*   ent_ids   = (const int*)d_in[28];
  const int*   ent_range = (const int*)d_in[29];

  const int Bb = 4, S = 2048, H = 1024, NH = 16, HD = 64, FFN = 4096, T = 50;
  const int M = Bb * S;                       // 8192 rows
  const size_t MB = (size_t)M * H;            // 8M elements

  float* ws   = (float*)d_ws;
  float* tF   = ws;                           // [M,H] f32 pre-LN sums
  float* a1F  = ws + MB;                      // [M,H] f32 (residual for t2)
  float* a2F  = ws + 2 * MB;                  // [M,H] f32 (residual for t3)
  __bf16* pool   = (__bf16*)(ws + 3 * MB);
  __bf16* xb     = pool;                      // [M,H]
  __bf16* Qb     = pool + MB;                 // Q, later Qc
  __bf16* Kb     = pool + 2 * MB;             // K, later a1 (bf16)
  __bf16* Vb     = pool + 3 * MB;             // V, later a2 (bf16)
  __bf16* ctxb   = pool + 4 * MB;             // attention outputs
  __bf16* interb = pool + 5 * MB;             // [M,FFN] = 4*MB elements
  __bf16* tagsb  = pool + 9 * MB;             // [T,H]
  __bf16* kcb    = tagsb + (size_t)T * H;
  __bf16* vcb    = kcb   + (size_t)T * H;

  const dim3 blk256(256), blk128(128);
  const dim3 gH((M + 127) / 128, H / 128);    // 64 x 8
  const dim3 gT(1, H / 128);                  // tag projections
  const dim3 gI((M + 127) / 128, FFN / 128);  // 64 x 32
  const dim3 gA(S / 64, Bb * NH);             // 32 x 64
  const int n4 = (int)(MB / 4);

  // --- self-attention ---
  cvt_bf16_kernel<<<dim3(n4 / 256), blk256, 0, stream>>>(x, xb, n4);
  gemm_kernel<0,true ><<<gH, blk256, 0, stream>>>(xb, sq_w, sq_b, nullptr, nullptr, Qb, M, H, H);
  gemm_kernel<0,true ><<<gH, blk256, 0, stream>>>(xb, sk_w, sk_b, nullptr, nullptr, Kb, M, H, H);
  gemm_kernel<0,true ><<<gH, blk256, 0, stream>>>(xb, sv_w, sv_b, nullptr, nullptr, Vb, M, H, H);
  flash_self_attn_kernel<<<gA, blk128, 0, stream>>>(Qb, Kb, Vb, ctxb, ent_range, Bb, S, NH, HD);
  gemm_kernel<2,false><<<gH, blk256, 0, stream>>>(ctxb, so_w, so_b, x, tF, nullptr, M, H, H);
  layernorm_kernel<<<dim3(M), blk256, 0, stream>>>(tF, sln_g, sln_b, a1F, Kb, H);   // a1

  // --- cross-attention onto entity tags ---
  gather_tags_kernel<<<dim3(T), blk256, 0, stream>>>(emb, ent_ids, tagsb, H);
  gemm_kernel<0,true ><<<gT, blk256, 0, stream>>>(tagsb, ck_w, ck_b, nullptr, nullptr, kcb, T, H, H);
  gemm_kernel<0,true ><<<gT, blk256, 0, stream>>>(tagsb, cv_w, cv_b, nullptr, nullptr, vcb, T, H, H);
  gemm_kernel<0,true ><<<gH, blk256, 0, stream>>>(Kb, cq_w, cq_b, nullptr, nullptr, Qb, M, H, H); // Qc
  cross_attn_kernel<<<gA, blk128, 0, stream>>>(Qb, kcb, vcb, ctxb, Bb, S, NH, HD, T);
  gemm_kernel<2,false><<<gH, blk256, 0, stream>>>(ctxb, co_w, co_b, a1F, tF, nullptr, M, H, H);
  layernorm_kernel<<<dim3(M), blk256, 0, stream>>>(tF, cln_g, cln_b, a2F, Vb, H);   // a2

  // --- FFN ---
  gemm_kernel<1,true ><<<gI, blk256, 0, stream>>>(Vb, i_w, i_b, nullptr, nullptr, interb, M, FFN, H);
  gemm_kernel<2,false><<<gH, blk256, 0, stream>>>(interb, o_w, o_b, a2F, tF, nullptr, M, H, FFN);
  layernorm_kernel<<<dim3(M), blk256, 0, stream>>>(tF, oln_g, oln_b, (float*)d_out, nullptr, H);
}
